// MoE_layer_flux_47150150975576
// MI455X (gfx1250) — compile-verified, matrix operand order checked by reading, not run on hardware
//
#include <hip/hip_runtime.h>
#include <hip/hip_bf16.h>

typedef _Float16 half_t;
typedef __attribute__((ext_vector_type(16))) _Float16 v16h;
typedef __attribute__((ext_vector_type(8)))  _Float16 v8h;
typedef __attribute__((ext_vector_type(4)))  _Float16 v4h;
typedef __attribute__((ext_vector_type(8)))  float    v8f;
typedef __attribute__((ext_vector_type(4)))  float    v4f;

// GCC-style int4 with address-space-qualified pointers, matching the
// parameter types of __builtin_amdgcn_global_load_async_to_lds_b128.
typedef int v4i_g __attribute__((vector_size(4 * sizeof(int))));
typedef __attribute__((address_space(1))) v4i_g* as1_v4i_p;   // global
typedef __attribute__((address_space(3))) v4i_g* as3_v4i_p;   // LDS

#define NTOKENS 8192
#define HDIM 1024
#define FFN 2048
#define NEXPERTS 64
#define TOPK 2
#define MSLOTS (NTOKENS * TOPK)          // 16384
#define PER_EXPERT (MSLOTS / NEXPERTS)   // 256

// GEMM tiling: one block covers a full expert's 256 rows (weights read once).
#define BM 256
#define BN 64
#define BK 32
#define APITCH (BK + 8)                  // LDS row pitch in halves (80B, conflict-reducing)

// CDNA5 async global->LDS path (guarded; falls back to register copies).
#if defined(__HIP_DEVICE_COMPILE__) && __has_builtin(__builtin_amdgcn_global_load_async_to_lds_b128) && __has_builtin(__builtin_amdgcn_s_wait_asynccnt)
#define HAVE_ASYNC 1
#else
#define HAVE_ASYNC 0
#endif

// ---------------------------------------------------------------------------
// Branch-free exact-GELU: erf via Abramowitz-Stegun 7.1.26 (|err| < 1.5e-7,
// far below the f16 storage rounding of the activation buffer). The rational
// term uses raw v_rcp_f32 (~1 ulp) instead of an IEEE divide - invisible at
// this accuracy level but ~10x fewer VALU ops.
// ---------------------------------------------------------------------------
__device__ __forceinline__ float gelu_f(float x) {
    const float ax = fabsf(x) * 0.70710678118654752440f;  // |x|/sqrt(2)
#if defined(__HIP_DEVICE_COMPILE__) && __has_builtin(__builtin_amdgcn_rcpf)
    const float t  = __builtin_amdgcn_rcpf(1.0f + 0.3275911f * ax);
#else
    const float t  = 1.0f / (1.0f + 0.3275911f * ax);
#endif
    const float poly = t * (0.254829592f +
                       t * (-0.284496736f +
                       t * (1.421413741f +
                       t * (-1.453152027f +
                       t * 1.061405429f))));
    const float e   = __expf(-ax * ax);
    const float erf_ax = 1.0f - poly * e;
    const float erfv = copysignf(erf_ax, x);
    return 0.5f * x * (1.0f + erfv);
}

// ---------------------------------------------------------------------------
// Kernel 1: scatter tokens into expert-sorted slots, fp32 -> f16.
// ---------------------------------------------------------------------------
__global__ void k_scatter(const float* __restrict__ in, const int* __restrict__ sidx,
                          half_t* __restrict__ X) {
    const int t   = blockIdx.x;
    const int tid = threadIdx.x;               // 0..255, H/4 = 256
    const float4 v = ((const float4*)(in + (size_t)t * HDIM))[tid];
    v4h h;
    h[0] = (half_t)v.x; h[1] = (half_t)v.y; h[2] = (half_t)v.z; h[3] = (half_t)v.w;
    const int s0 = sidx[t * TOPK + 0];
    const int s1 = sidx[t * TOPK + 1];
    ((v4h*)(X + (size_t)s0 * HDIM))[tid] = h;
    ((v4h*)(X + (size_t)s1 * HDIM))[tid] = h;
}

// ---------------------------------------------------------------------------
// Grouped GEMM: per expert e, Out[256 x LDN] = A[256 x KDIM] (f16) * W_e[KDIM x LDN] (f32).
// Double-buffered LDS, one barrier per K-step. A tile goes global->LDS via the
// CDNA5 async path; B tile is prefetched to VGPRs, converted f32->f16 and stored
// K-transposed so both fragments are contiguous ds_load_b128's.
// 256 threads = 8 waves; each wave computes a 32x64 sub-tile = 2x4 WMMA accums.
// ---------------------------------------------------------------------------
template <int KDIM, int LDN, bool DO_GELU>
__global__ void __launch_bounds__(256)
k_gemm(const half_t* __restrict__ A, const float* __restrict__ W, void* __restrict__ Out) {
    __shared__ half_t lds_a[2][BM * APITCH];   // 2 x 20.0 KB
    __shared__ half_t lds_b[2][BN * APITCH];   // 2 x  5.0 KB  (stored as BT[n][k])

    const int ntiles = LDN / BN;
    const int e      = blockIdx.x / ntiles;
    const int n0     = (blockIdx.x % ntiles) * BN;
    const int tid    = threadIdx.x;
    const int lane   = tid & 31;
    const int wv     = tid >> 5;              // wave id 0..7
    const int hl     = lane & 15;             // low lane index
    const int ks     = (lane >> 4) * 8;       // K-half select per WMMA 16-bit layout

    const size_t arow0 = (size_t)e * PER_EXPERT;
    const float* Wb    = W + (size_t)e * KDIM * LDN;

    // B-tile per-thread geometry: 32(K) x 64(N) floats, 2 x float4 per thread.
    const int nq = (tid & 15) * 4;            // 0..60
    const int kk = (tid >> 4) * 2;            // 0..30

    // A tile: issue per-thread global->LDS copies (async when available).
    auto copy_a_tile = [&](int kb, int buf) {
#pragma unroll
        for (int i = 0; i < 4; ++i) {
            const int idx = tid + 256 * i;
            const int row = idx >> 2;          // 0..255
            const int seg = idx & 3;           // 8-half segment
            const half_t* g = A + (arow0 + row) * (size_t)KDIM + kb + seg * 8;
            half_t* l = &lds_a[buf][row * APITCH + seg * 8];
#if HAVE_ASYNC
            __builtin_amdgcn_global_load_async_to_lds_b128(
                (as1_v4i_p)g, (as3_v4i_p)l, 0, 0);
#else
            *(v8h*)l = *(const v8h*)g;
#endif
        }
    };
    auto load_b_regs = [&](int kb, v4f breg[2]) {
#pragma unroll
        for (int j = 0; j < 2; ++j)
            breg[j] = *(const v4f*)(Wb + (size_t)(kb + kk + j) * LDN + n0 + nq);
    };
    auto store_b_lds = [&](int buf, const v4f breg[2]) {
#pragma unroll
        for (int j = 0; j < 2; ++j)
#pragma unroll
            for (int q = 0; q < 4; ++q)
                lds_b[buf][(nq + q) * APITCH + kk + j] = (half_t)breg[j][q];
    };

    v8f acc[2][4];
#pragma unroll
    for (int mt = 0; mt < 2; ++mt)
#pragma unroll
        for (int nt = 0; nt < 4; ++nt) {
            v8f z = {0.f, 0.f, 0.f, 0.f, 0.f, 0.f, 0.f, 0.f};
            acc[mt][nt] = z;
        }

    // Prologue: stage tile 0 into buffer 0.
    v4f breg[2];
    copy_a_tile(0, 0);
    load_b_regs(0, breg);
    store_b_lds(0, breg);
#if HAVE_ASYNC
    __builtin_amdgcn_s_wait_asynccnt(0);
#endif
    __syncthreads();

    const int NK = KDIM / BK;
    for (int it = 0; it < NK; ++it) {
        const int cur = it & 1;
        const int nxt = cur ^ 1;
        const bool more = (it + 1) < NK;
        if (more) {
            copy_a_tile((it + 1) * BK, nxt);   // async: overlaps with WMMAs below
            load_b_regs((it + 1) * BK, breg);
        }

        // --- fragments (ISA 16-bit 16x32 A layout: K = {ks..ks+7, ks+16..ks+23}) ---
        v16h afrag[2], bfrag[4];
#pragma unroll
        for (int mt = 0; mt < 2; ++mt) {
            const half_t* base = &lds_a[cur][(wv * 32 + mt * 16 + hl) * APITCH];
            v8h lo = *(const v8h*)(base + ks);
            v8h hi = *(const v8h*)(base + 16 + ks);
#pragma unroll
            for (int i = 0; i < 8; ++i) { afrag[mt][i] = lo[i]; afrag[mt][i + 8] = hi[i]; }
        }
#pragma unroll
        for (int nt = 0; nt < 4; ++nt) {
            const half_t* base = &lds_b[cur][(nt * 16 + hl) * APITCH];
            v8h lo = *(const v8h*)(base + ks);
            v8h hi = *(const v8h*)(base + 16 + ks);
#pragma unroll
            for (int i = 0; i < 8; ++i) { bfrag[nt][i] = lo[i]; bfrag[nt][i + 8] = hi[i]; }
        }
        // --- 8 WMMAs per wave per K-step ---
#pragma unroll
        for (int mt = 0; mt < 2; ++mt)
#pragma unroll
            for (int nt = 0; nt < 4; ++nt)
                acc[mt][nt] = __builtin_amdgcn_wmma_f32_16x16x32_f16(
                    false, afrag[mt], false, bfrag[nt], (short)0, acc[mt][nt], false, false);

        if (more) {
            store_b_lds(nxt, breg);
#if HAVE_ASYNC
            __builtin_amdgcn_s_wait_asynccnt(0);
#endif
            __syncthreads();
        }
    }

    // --- epilogue: C/D layout vgpr r -> row (r + 8*(lane>=16)), col = lane%16 ---
    const int    col0 = n0 + hl;
    const size_t rowb = arow0 + wv * 32 + (lane >> 4) * 8;
#pragma unroll
    for (int mt = 0; mt < 2; ++mt) {
#pragma unroll
        for (int nt = 0; nt < 4; ++nt) {
#pragma unroll
            for (int r = 0; r < 8; ++r) {
                const size_t row = rowb + mt * 16 + r;
                float v = acc[mt][nt][r];
                if constexpr (DO_GELU) {
                    ((half_t*)Out)[row * (size_t)LDN + col0 + nt * 16] = (half_t)gelu_f(v);
                } else {
                    ((float*)Out)[row * (size_t)LDN + col0 + nt * 16] = v;
                }
            }
        }
    }
}

// ---------------------------------------------------------------------------
// Kernel 4: gather top-2 partials per token and reduce.
// ---------------------------------------------------------------------------
__global__ void k_gather(const float* __restrict__ h2, const int* __restrict__ sidx,
                         float* __restrict__ out) {
    const int t   = blockIdx.x;
    const int tid = threadIdx.x;               // H/4 = 256
    const int s0  = sidx[t * TOPK + 0];
    const int s1  = sidx[t * TOPK + 1];
    const float4 a = ((const float4*)(h2 + (size_t)s0 * HDIM))[tid];
    const float4 b = ((const float4*)(h2 + (size_t)s1 * HDIM))[tid];
    float4 r;
    r.x = a.x + b.x; r.y = a.y + b.y; r.z = a.z + b.z; r.w = a.w + b.w;
    ((float4*)(out + (size_t)t * HDIM))[tid] = r;
}

extern "C" void kernel_launch(void* const* d_in, const int* in_sizes, int n_in,
                              void* d_out, int out_size, void* d_ws, size_t ws_size,
                              hipStream_t stream) {
    const float* inputs = (const float*)d_in[0];   // [8192, 1024] f32
    const float* w1     = (const float*)d_in[1];   // [64, 1024, 2048] f32
    const float* w2     = (const float*)d_in[2];   // [64, 2048, 1024] f32
    const int*   sidx   = (const int*)d_in[3];     // [8192, 2] int32
    float*       out    = (float*)d_out;           // [8192, 1024] f32

    char* ws = (char*)d_ws;
    half_t* X   = (half_t*)ws;                                              // 32 MB
    half_t* act = (half_t*)(ws + (size_t)MSLOTS * HDIM * sizeof(half_t));   // 64 MB
    float*  h2  = (float*)(ws + (size_t)MSLOTS * HDIM * sizeof(half_t)
                              + (size_t)MSLOTS * FFN * sizeof(half_t));     // 64 MB

    // 1) scatter + convert to f16
    k_scatter<<<NTOKENS, 256, 0, stream>>>(inputs, sidx, X);
    // 2) grouped GEMM1 (H -> FFN) + exact GELU
    k_gemm<HDIM, FFN, true><<<NEXPERTS * (FFN / BN), 256, 0, stream>>>(X, w1, act);
    // 3) grouped GEMM2 (FFN -> H)
    k_gemm<FFN, HDIM, false><<<NEXPERTS * (HDIM / BN), 256, 0, stream>>>(act, w2, h2);
    // 4) gather top-2 and sum
    k_gather<<<NTOKENS, 256, 0, stream>>>(h2, sidx, out);
}
